// DARTSCell_12292196401475
// MI455X (gfx1250) — matrix operand order repaired and checked by reading, not compiled, not run on hardware
//
#include <hip/hip_runtime.h>
#include <stdint.h>

#define T_STEPS 1024
#define BATCH   64
#define HID     512
#define BH      (BATCH * HID)      /* 32768 elements */
#define NWG     32
#define STAGE_BYTES 32768          /* one weight stage: 2 n-tiles x 16 k-tiles x 1KB frag */

typedef __attribute__((ext_vector_type(16))) __bf16   v16bf;
typedef __attribute__((ext_vector_type(8)))  float    v8f;
typedef __attribute__((ext_vector_type(4)))  unsigned v4u;
typedef __attribute__((ext_vector_type(8)))  int      v8i;
typedef __attribute__((ext_vector_type(4)))  int      v4i;

// ---------------- helpers ----------------

__device__ __forceinline__ unsigned short f2bf(float f) {
  unsigned u = __float_as_uint(f);
  unsigned r = u + 0x7FFFu + ((u >> 16) & 1u);   // round-to-nearest-even
  return (unsigned short)(r >> 16);
}

__device__ __forceinline__ float sigm(float x) { return 1.0f / (1.0f + __expf(-x)); }
__device__ __forceinline__ float tanh_(float x) {
  float e = __expf(-2.0f * x);
  return (1.0f - e) / (1.0f + e);
}
__device__ __forceinline__ float actfn(int a, float x) {
  if (a == 0) return sigm(x);
  if (a == 1) return fmaxf(x, 0.0f);
  if (a == 2) return x;
  return tanh_(x);
}

// Monotonic cross-workgroup barrier (persistent grid of NWG workgroups).
__device__ __forceinline__ void global_barrier(unsigned* cnt, unsigned target) {
  __threadfence();
  __syncthreads();
  if (threadIdx.x == 0) {
    __hip_atomic_fetch_add(cnt, 1u, __ATOMIC_RELEASE, __HIP_MEMORY_SCOPE_AGENT);
    while (__hip_atomic_load(cnt, __ATOMIC_ACQUIRE, __HIP_MEMORY_SCOPE_AGENT) < target) {
      __builtin_amdgcn_s_sleep(1);
    }
  }
  __syncthreads();
  __threadfence();
}

// TDM: async copy of one contiguous 32KB weight stage (global -> LDS).
// 1-D tensor descriptor, data_size = 8B, 4096 units. Issued by wave 0 only.
__device__ __forceinline__ void tdm_stage_load(const void* src, unsigned ldsOff) {
  uint64_t ga = (uint64_t)(uintptr_t)src;
  v4u g0;
  g0[0] = 1u;                                              // count=1, user mode
  g0[1] = ldsOff;                                          // lds_addr
  g0[2] = (unsigned)(ga & 0xFFFFFFFFu);                    // global_addr[31:0]
  g0[3] = (unsigned)((ga >> 32) & 0x01FFFFFFu) | (2u << 30); // addr[56:32] | type=2
  v8i g1;
  g1[0] = (int)(3u << 16);                 // workgroup_mask=0, data_size=3 (8B)
  g1[1] = (int)((4096u & 0xFFFFu) << 16);  // tensor_dim0 lo16
  g1[2] = (int)((4096u >> 16) | (1u << 16)); // tensor_dim0 hi16 | tensor_dim1=1 lo
  g1[3] = (int)(4096u << 16);              // tile_dim0 = 4096 units
  g1[4] = 0;                               // tile_dim1/2 unused
  g1[5] = (int)4096u;                      // tensor_dim0_stride lo32
  g1[6] = 0;
  g1[7] = 0;
  v4i z4 = {0, 0, 0, 0};
#if defined(__clang_major__) && (__clang_major__ >= 23)
  v8i z8 = {0, 0, 0, 0, 0, 0, 0, 0};
  __builtin_amdgcn_tensor_load_to_lds(g0, g1, z4, z4, z8, 0);
#else
  __builtin_amdgcn_tensor_load_to_lds(g0, g1, z4, z4, 0);
#endif
}

// One weight stage GEMM slice: 16 k-tiles, 2 n-tiles (c-half, h-half).
// A fragments (16x32 bf16) loaded from row-major bf16 matrix (stride 512 elem),
// abase already adjusted per lane: + row*1024B + (lane>=16 ? 32B : 0).
// B fragments come from LDS in fragment-ready layout: [j][kt][lane*32B].
union AFrag { v16bf v; uint4 q[2]; };

__device__ __forceinline__ void gemm_stage(const char* __restrict__ abase,
                                           const char* __restrict__ sm,
                                           int lane, v8f& cacc, v8f& hacc) {
#pragma unroll
  for (int kt = 0; kt < 16; ++kt) {
    AFrag a;
    a.q[0] = *(const uint4*)(abase + kt * 64);        // K k0..k0+7
    a.q[1] = *(const uint4*)(abase + kt * 64 + 16);   // K k0+8..k0+15
    v16bf b0 = *(const v16bf*)(sm + kt * 1024 + lane * 32);
    v16bf b1 = *(const v16bf*)(sm + 16384 + kt * 1024 + lane * 32);
    cacc = __builtin_amdgcn_wmma_f32_16x16x32_bf16(false, a.v, false, b0,
                                                   (short)0, cacc, false, false);
    hacc = __builtin_amdgcn_wmma_f32_16x16x32_bf16(false, a.v, false, b1,
                                                   (short)0, hacc, false, false);
  }
}

// ---------------- per-node (templated so all state indices are constants) ----

template <int I>
__device__ __forceinline__ void darts_node(
    v8f (&st)[9], unsigned short* __restrict__ statesbf,
    const char* wsf_wg, const char* w0src, char* smem, int& p,
    unsigned& bar_target, unsigned* barcnt,
    int wave, int lane, int aoff, int mb, int col) {
  constexpr int  PRED_[8] = {0, 1, 1, 1, 2, 5, 3, 5};
  constexpr int  ACT_[8]  = {0, 1, 1, 2, 3, 0, 3, 1};   // 0=sig 1=relu 2=id 3=tanh
  constexpr bool BAR_[8]  = {true, true, false, false, true, true, false, false};
  constexpr bool WST_[9]  = {true, true, true, true, false, true, false, false, false};

  if (BAR_[I]) { bar_target += NWG; global_barrier(barcnt, bar_target); }

  if (wave == 0) {   // prefetch next stage's weights into the other LDS buffer
    const char* nsrc = (I == 7) ? w0src
                                : (wsf_wg + (size_t)(I + 1) * ((size_t)NWG * STAGE_BYTES));
    tdm_stage_load(nsrc, (unsigned)(uintptr_t)(smem + (p ^ 1) * STAGE_BYTES));
  }

  v8f ca = {0, 0, 0, 0, 0, 0, 0, 0}, ha = {0, 0, 0, 0, 0, 0, 0, 0};
  const char* A = (const char*)(statesbf + PRED_[I] * BH) + aoff;
  gemm_stage(A, smem + p * STAGE_BYTES, lane, ca, ha);

  if (wave == 0) __builtin_amdgcn_s_wait_tensorcnt(0);
  __syncthreads();
  p ^= 1;

#pragma unroll
  for (int r = 0; r < 8; ++r) {
    float c  = sigm(ca[r]);
    float h  = actfn(ACT_[I], ha[r]);
    float sp = st[PRED_[I]][r];
    st[I + 1][r] = sp + c * (h - sp);
  }
  if (WST_[I + 1]) {
#pragma unroll
    for (int r = 0; r < 8; ++r)
      statesbf[(size_t)(I + 1) * BH + (mb + r) * HID + col] = f2bf(st[I + 1][r]);
  }
}

// ---------------- persistent cell kernel ----------------

__global__ __launch_bounds__(128, 1) void darts_cell_kernel(
    const unsigned short* __restrict__ xbf,   // T*B*H bf16
    const unsigned short* __restrict__ w0f,   // fragment-ready W0 (per-wg 64KB blocks)
    const unsigned short* __restrict__ wsf,   // fragment-ready Ws (per node/wg 32KB)
    unsigned short* __restrict__ hprevbf,     // B*H bf16
    unsigned short* __restrict__ statesbf,    // 9*B*H bf16
    float* __restrict__ out,                  // T*B*H + B*H f32
    unsigned* __restrict__ barcnt) {
  __shared__ __align__(128) char smem[2 * STAGE_BYTES];

  const int wg   = (int)blockIdx.x;            // 0..31, owns hidden cols [16wg,16wg+16)
  const int lane = (int)(threadIdx.x & 31);
  const int wave = (int)(threadIdx.x >> 5);    // M-tile 0..3
  const int nlo  = lane & 15;
  const int row  = wave * 16 + nlo;                       // A-fragment row
  const int mb   = wave * 16 + ((lane >> 4) << 3);        // C/D row base
  const int col  = wg * 16 + nlo;                         // owned state column
  const int aoff = row * 1024 + ((lane >> 4) << 5);       // per-lane A byte offset

  const char* w0src  = (const char*)w0f + (size_t)wg * 65536;
  const char* wsf_wg = (const char*)wsf + (size_t)wg * STAGE_BYTES;

  if (wave == 0) {                       // prologue: stage W0 half0 into buf0
    tdm_stage_load(w0src, (unsigned)(uintptr_t)smem);
    __builtin_amdgcn_s_wait_tensorcnt(0);
  }
  __syncthreads();

  int p = 0;
  unsigned bar_target = 0;
  v8f hprev = {0, 0, 0, 0, 0, 0, 0, 0};
  v8f st[9];

  for (int t = 0; t < T_STEPS; ++t) {
    if (t + 1 < T_STEPS)
      __builtin_prefetch(xbf + (size_t)(t + 1) * BH + row * HID, 0, 1);

    v8f cacc = {0, 0, 0, 0, 0, 0, 0, 0}, hacc = {0, 0, 0, 0, 0, 0, 0, 0};

    // --- W0 stage A: K = x_t (cols 0..511 of [x,h]) ---
    if (wave == 0)
      tdm_stage_load(w0src + STAGE_BYTES,
                     (unsigned)(uintptr_t)(smem + (p ^ 1) * STAGE_BYTES));
    gemm_stage((const char*)(xbf + (size_t)t * BH) + aoff,
               smem + p * STAGE_BYTES, lane, cacc, hacc);
    if (wave == 0) __builtin_amdgcn_s_wait_tensorcnt(0);
    __syncthreads();
    p ^= 1;

    // --- W0 stage B: K = h_prev (cols 512..1023), accumulators carry over ---
    if (wave == 0)
      tdm_stage_load(wsf_wg, (unsigned)(uintptr_t)(smem + (p ^ 1) * STAGE_BYTES));
    gemm_stage((const char*)hprevbf + aoff, smem + p * STAGE_BYTES, lane, cacc, hacc);
    if (wave == 0) __builtin_amdgcn_s_wait_tensorcnt(0);
    __syncthreads();
    p ^= 1;

    // state0 = h_prev + sigmoid(c0) * (tanh(h0) - h_prev)
#pragma unroll
    for (int r = 0; r < 8; ++r) {
      float c0 = sigm(cacc[r]);
      float h0 = tanh_(hacc[r]);
      st[0][r] = hprev[r] + c0 * (h0 - hprev[r]);
    }
#pragma unroll
    for (int r = 0; r < 8; ++r)
      statesbf[(mb + r) * HID + col] = f2bf(st[0][r]);

    darts_node<0>(st, statesbf, wsf_wg, w0src, smem, p, bar_target, barcnt, wave, lane, aoff, mb, col);
    darts_node<1>(st, statesbf, wsf_wg, w0src, smem, p, bar_target, barcnt, wave, lane, aoff, mb, col);
    darts_node<2>(st, statesbf, wsf_wg, w0src, smem, p, bar_target, barcnt, wave, lane, aoff, mb, col);
    darts_node<3>(st, statesbf, wsf_wg, w0src, smem, p, bar_target, barcnt, wave, lane, aoff, mb, col);
    darts_node<4>(st, statesbf, wsf_wg, w0src, smem, p, bar_target, barcnt, wave, lane, aoff, mb, col);
    darts_node<5>(st, statesbf, wsf_wg, w0src, smem, p, bar_target, barcnt, wave, lane, aoff, mb, col);
    darts_node<6>(st, statesbf, wsf_wg, w0src, smem, p, bar_target, barcnt, wave, lane, aoff, mb, col);
    darts_node<7>(st, statesbf, wsf_wg, w0src, smem, p, bar_target, barcnt, wave, lane, aoff, mb, col);

    // h_new = mean(states[1..8])
#pragma unroll
    for (int r = 0; r < 8; ++r) {
      float s = st[1][r] + st[2][r] + st[3][r] + st[4][r] +
                st[5][r] + st[6][r] + st[7][r] + st[8][r];
      hprev[r] = s * 0.125f;
    }
    float* o = out + (size_t)t * BH;
#pragma unroll
    for (int r = 0; r < 8; ++r) o[(mb + r) * HID + col] = hprev[r];
    if (t == T_STEPS - 1) {
      float* o2 = out + (size_t)T_STEPS * BH;
#pragma unroll
      for (int r = 0; r < 8; ++r) o2[(mb + r) * HID + col] = hprev[r];
    }
#pragma unroll
    for (int r = 0; r < 8; ++r) hprevbf[(mb + r) * HID + col] = f2bf(hprev[r]);

    bar_target += NWG;
    global_barrier(barcnt, bar_target);   // next step's W0b reads hprevbf
  }
}

// ---------------- prep kernels ----------------

__global__ void prep_inputs_kernel(const float* __restrict__ x,
                                   unsigned short* __restrict__ xbf, long long n) {
  long long i = (long long)blockIdx.x * blockDim.x + threadIdx.x;
  long long stride = (long long)gridDim.x * blockDim.x;
  for (; i < n; i += stride) xbf[i] = f2bf(x[i]);
}

__global__ void prep_misc_kernel(unsigned short* __restrict__ hprevbf,
                                 unsigned* __restrict__ barcnt, int n) {
  int i = (int)(blockIdx.x * blockDim.x + threadIdx.x);
  int stride = (int)(gridDim.x * blockDim.x);
  for (int j = i; j < n; j += stride) hprevbf[j] = 0;
  if (i == 0) *barcnt = 0u;
}

// Shuffle weights into WMMA-B fragment-ready bf16 layout.
// Fragment = 1KB: lane l (0..31) holds 16 bf16 at frag + l*32:
//   n = colbase + (l & 15);  k = kbase + e + ((l >> 4) << 4)   (e = 0..15)
// W0 layout:  per wg (64KB): [khalf(2)][j(2)][ktl(16)] frags, wg = frag>>6
// Ws layout:  per node: [wg(32)][j(2)][ktl(16)] frags
__global__ void prep_weights_kernel(const float* __restrict__ W0,
                                    const float* __restrict__ Ws,
                                    unsigned short* __restrict__ w0f,
                                    unsigned short* __restrict__ wsf) {
  const long long NFRAG_W0 = 2048;
  const long long NFRAG_WS = 8192;
  const long long total = (NFRAG_W0 + NFRAG_WS) * 512;
  long long g = (long long)blockIdx.x * blockDim.x + threadIdx.x;
  long long stride = (long long)gridDim.x * blockDim.x;
  for (; g < total; g += stride) {
    long long frag = g >> 9;
    int within = (int)(g & 511);
    int l = within >> 4, e = within & 15;
    if (frag < NFRAG_W0) {
      int wg = (int)(frag >> 6), rem = (int)(frag & 63);
      int h = rem >> 5, j = (rem >> 4) & 1, ktl = rem & 15;
      int kb = (h * 16 + ktl) * 32;
      int n_ = j * 512 + wg * 16 + (l & 15);
      int k_ = kb + e + ((l >> 4) << 4);
      w0f[frag * 512 + l * 16 + e] = f2bf(W0[(long long)k_ * 1024 + n_]);
    } else {
      long long fi = frag - NFRAG_W0;
      int node = (int)(fi >> 10), rem = (int)(fi & 1023);
      int wg = rem >> 5, rem2 = rem & 31;
      int j = rem2 >> 4, ktl = rem2 & 15;
      int kb = ktl * 32;
      int n_ = j * 512 + wg * 16 + (l & 15);
      int k_ = kb + e + ((l >> 4) << 4);
      wsf[fi * 512 + l * 16 + e] =
          f2bf(Ws[(long long)node * 512 * 1024 + (long long)k_ * 1024 + n_]);
    }
  }
}

// ---------------- launch ----------------

extern "C" void kernel_launch(void* const* d_in, const int* in_sizes, int n_in,
                              void* d_out, int out_size, void* d_ws, size_t ws_size,
                              hipStream_t stream) {
  (void)in_sizes; (void)n_in; (void)out_size; (void)ws_size;
  const float* x  = (const float*)d_in[0];   // (T,B,H)
  const float* W0 = (const float*)d_in[1];   // (2H,2H)
  const float* Ws = (const float*)d_in[2];   // (8,H,2H)
  float* out = (float*)d_out;

  char* ws = (char*)d_ws;
  size_t off = 0;
  unsigned short* xbf = (unsigned short*)(ws + off); off += (size_t)T_STEPS * BH * 2;
  unsigned short* w0f = (unsigned short*)(ws + off); off += (size_t)1024 * 1024 * 2;
  unsigned short* wsf = (unsigned short*)(ws + off); off += (size_t)8 * 512 * 1024 * 2;
  unsigned short* hprevbf = (unsigned short*)(ws + off); off += (size_t)BH * 2;
  unsigned short* statesbf = (unsigned short*)(ws + off); off += (size_t)9 * BH * 2;
  unsigned* barcnt = (unsigned*)(ws + off);

  prep_inputs_kernel<<<4096, 256, 0, stream>>>(x, xbf, (long long)T_STEPS * BH);
  prep_weights_kernel<<<2048, 256, 0, stream>>>(W0, Ws, w0f, wsf);
  prep_misc_kernel<<<64, 256, 0, stream>>>(hprevbf, barcnt, BH);
  darts_cell_kernel<<<NWG, 128, 0, stream>>>(xbf, w0f, wsf, hprevbf, statesbf, out, barcnt);
}
